// Attention_90417651516187
// MI455X (gfx1250) — matrix operand order repaired and checked by reading, not compiled, not run on hardware
//
#include <hip/hip_runtime.h>
#include <hip/hip_bf16.h>

#define DIMN     2048
#define HEADS    16
#define HEAD_DIM 128
#define SEQ      2048
#define BATCH    2

typedef __bf16 bf16;
typedef __attribute__((ext_vector_type(16))) __bf16 v16bf;
typedef __attribute__((ext_vector_type(4)))  __bf16 v4bf;
typedef __attribute__((ext_vector_type(8)))  float  v8f;

union FragAB {            // one 16x32 (A) or 32x16 (B) bf16 WMMA operand: 32 bytes/lane
    v16bf v;
    uint4 q[2];
};

__device__ __forceinline__ v8f wmma_bf16(const FragAB& a, const FragAB& b, v8f c) {
    return __builtin_amdgcn_wmma_f32_16x16x32_bf16(
        /*neg_a=*/false, a.v, /*neg_b=*/false, b.v,
        /*c_mod=*/(short)0, c, /*reuse_a=*/false, /*reuse_b=*/false);
}

// LDS byte address = low 32 bits of the generic address
__device__ __forceinline__ unsigned int lds_addr32(const void* p) {
    return (unsigned int)(unsigned long long)p;
}

// CDNA5 async copy: 16 bytes per lane, global -> LDS, tracked by ASYNCcnt
__device__ __forceinline__ void async_cp16(unsigned int lds_off, const bf16* g) {
    asm volatile("global_load_async_to_lds_b128 %0, %1, off"
                 :: "v"(lds_off), "v"((unsigned long long)g)
                 : "memory");
}
__device__ __forceinline__ void wait_async0() {
    asm volatile("s_wait_asynccnt 0x0" ::: "memory");
}
__device__ __forceinline__ void wait_ds0() {
    asm volatile("s_wait_dscnt 0x0" ::: "memory");
}

// A-fragment 16x32 from an LDS tile (row stride `stride` halves), k-run starts at k0
__device__ __forceinline__ FragAB lds_fragA(const bf16* base, int stride, int row0, int k0, int lane) {
    FragAB f;
    const bf16* p = base + (row0 + (lane & 15)) * stride + k0 + (lane >> 4) * 8;
    f.q[0] = *(const uint4*)(p);
    f.q[1] = *(const uint4*)(p + 16);
    return f;
}
// B-fragment 32x16 from an LDS tile (rows = N, contiguous K), k-run starts at k0
__device__ __forceinline__ FragAB lds_fragB(const bf16* base, int stride, int row0, int k0, int lane) {
    FragAB f;
    const bf16* p = base + (row0 + (lane & 15)) * stride + k0 + (lane >> 4) * 8;
    f.q[0] = *(const uint4*)(p);
    f.q[1] = *(const uint4*)(p + 16);
    return f;
}

// ---------------- cast fp32 -> bf16 ----------------
__global__ void cast_f32_bf16(const float* __restrict__ in, bf16* __restrict__ out, int n) {
    int idx = (blockIdx.x * blockDim.x + threadIdx.x) * 4;
    if (idx < n) {
        float4 f = *(const float4*)(in + idx);
        v4bf o = { (bf16)f.x, (bf16)f.y, (bf16)f.z, (bf16)f.w };
        *(v4bf*)(out + idx) = o;
    }
}

// ---------------- GEMM: C[M,N] = A[M,K] * W[N,K]^T ----------------
// Block tile 64x256 (8 waves, 2x4; wave tile 32x64), k-step 64,
// async double-buffered LDS staging. 16 WMMAs per wave per barrier.
// mode 0: fp32 row-major store to Cf   (output projection)
// mode 1: bf16 store as [B,H,S,D]      (Q, K)
// mode 2: bf16 store as [B,H,D,S]      (V transposed)
__global__ void gemm_xwT(const bf16* __restrict__ A, const bf16* __restrict__ W,
                         float* __restrict__ Cf, bf16* __restrict__ Cb, int mode) {
    __shared__ bf16 sA[2][64 * 64];      // 2 x 8KB
    __shared__ bf16 sW[2][256 * 64];     // 2 x 32KB

    const int t    = threadIdx.x;
    const int lane = t & 31;
    const int wave = t >> 5;
    const int nblk = DIMN / 256;                   // 8 N-blocks
    const int m0   = (blockIdx.x / nblk) * 64;
    const int n0   = (blockIdx.x % nblk) * 256;
    const int mw   = (wave >> 2) * 32;             // wave row offset inside A tile
    const int nw   = (wave & 3) * 64;              // wave col offset inside W tile

    auto stage = [&](int buf, int k0) {
#pragma unroll
        for (int i = 0; i < 2; ++i) {              // A tile: 64 rows x 128B = 512 chunks
            const int id = t + i * 256;
            const int r = id >> 3, c = id & 7;
            async_cp16(lds_addr32(&sA[buf][r * 64 + c * 8]),
                       A + (size_t)(m0 + r) * DIMN + k0 + c * 8);
        }
#pragma unroll
        for (int i = 0; i < 8; ++i) {              // W tile: 256 rows x 128B = 2048 chunks
            const int id = t + i * 256;
            const int r = id >> 3, c = id & 7;
            async_cp16(lds_addr32(&sW[buf][r * 64 + c * 8]),
                       W + (size_t)(n0 + r) * DIMN + k0 + c * 8);
        }
    };

    v8f acc[2][4] = {};
    stage(0, 0);
    wait_async0();
    __syncthreads();

    int buf = 0;
    for (int k0 = 0; k0 < DIMN; k0 += 64) {
        if (k0 + 64 < DIMN) stage(buf ^ 1, k0 + 64);   // prefetch next tiles (async)

#pragma unroll
        for (int kk = 0; kk < 2; ++kk) {
            // batch-load all fragments of this k-chunk, then burst the WMMAs
            FragAB a0 = lds_fragA(sA[buf], 64, mw,      kk * 32, lane);
            FragAB a1 = lds_fragA(sA[buf], 64, mw + 16, kk * 32, lane);
            FragAB bf[4];
#pragma unroll
            for (int tt = 0; tt < 4; ++tt)
                bf[tt] = lds_fragB(sW[buf], 64, nw + 16 * tt, kk * 32, lane);
#pragma unroll
            for (int tt = 0; tt < 4; ++tt) {
                acc[0][tt] = wmma_bf16(a0, bf[tt], acc[0][tt]);
                acc[1][tt] = wmma_bf16(a1, bf[tt], acc[1][tt]);
            }
        }

        wait_async0();
        __syncthreads();
        buf ^= 1;
    }

    const int mrow = (lane >> 4) * 8;   // C layout: VGPR r, lanes>=16 -> M = r + 8
    const int col  = lane & 15;
#pragma unroll
    for (int ms = 0; ms < 2; ++ms) {
#pragma unroll
        for (int tt = 0; tt < 4; ++tt) {
#pragma unroll
            for (int r = 0; r < 8; ++r) {
                const int m = m0 + mw + ms * 16 + mrow + r;
                const int n = n0 + nw + 16 * tt + col;
                const float val = acc[ms][tt][r];
                if (mode == 0) {
                    Cf[(size_t)m * DIMN + n] = val;
                } else {
                    const int b = m / SEQ, s = m % SEQ;
                    const int h = n / HEAD_DIM, d = n % HEAD_DIM;
                    if (mode == 1)
                        Cb[(((size_t)(b * HEADS + h)) * SEQ + s) * HEAD_DIM + d] = (bf16)val;
                    else
                        Cb[(((size_t)(b * HEADS + h)) * HEAD_DIM + d) * SEQ + s] = (bf16)val;
                }
            }
        }
    }
}

// ---------------- fused flash attention ----------------
// 128-thread block = 4 waves = 4 consecutive 16-row q tiles of the SAME (b,h).
// K/V tiles staged to LDS with async copies (double buffered), shared by all 4 waves.
__global__ void flash_attn(const bf16* __restrict__ Q, const bf16* __restrict__ Km,
                           const bf16* __restrict__ Vt, const float* __restrict__ mask,
                           bf16* __restrict__ O) {
    __shared__ bf16 sK[2][32 * HEAD_DIM];   // 2 x 8KB : rows = kv (32), contiguous d
    __shared__ bf16 sV[2][HEAD_DIM * 32];   // 2 x 8KB : rows = d (128), contiguous kv
    __shared__ bf16 sP[4][16 * 32];         // per-wave P tile

    const int t    = threadIdx.x;
    const int lane = t & 31;
    const int wave = t >> 5;
    const int task0 = blockIdx.x * 4;            // 4 tasks per block, same (b,h)
    const int h  = (task0 >> 7) & (HEADS - 1);
    const int b  = task0 >> 11;
    const int qt = (task0 + wave) & (SEQ / 16 - 1);
    const int m0 = qt * 16;

    const bf16* Qh = Q  + ((size_t)(b * HEADS + h)) * SEQ * HEAD_DIM;
    const bf16* Kh = Km + ((size_t)(b * HEADS + h)) * SEQ * HEAD_DIM;
    const bf16* Vh = Vt + ((size_t)(b * HEADS + h)) * HEAD_DIM * SEQ;
    bf16* pshare = sP[wave];

    auto stage = [&](int buf, int kv0) {
#pragma unroll
        for (int i = 0; i < 4; ++i) {            // K tile: 32 rows x 256B = 512 chunks
            const int id = t + i * 128;
            const int r = id >> 4, c = id & 15;
            async_cp16(lds_addr32(&sK[buf][r * HEAD_DIM + c * 8]),
                       Kh + (size_t)(kv0 + r) * HEAD_DIM + c * 8);
        }
#pragma unroll
        for (int i = 0; i < 4; ++i) {            // V tile: 128 rows x 64B = 512 chunks
            const int id = t + i * 128;
            const int r = id >> 2, c = id & 3;
            async_cp16(lds_addr32(&sV[buf][r * 32 + c * 8]),
                       Vh + (size_t)r * SEQ + kv0 + c * 8);
        }
    };

    // Q fragments (per wave, loaded once from global)
    FragAB qf[4];
#pragma unroll
    for (int kk = 0; kk < 4; ++kk) {
        const bf16* p = Qh + (size_t)(m0 + (lane & 15)) * HEAD_DIM + kk * 32 + (lane >> 4) * 8;
        qf[kk].q[0] = *(const uint4*)(p);
        qf[kk].q[1] = *(const uint4*)(p + 16);
    }

    const float scale = 0.08838834764831845f;    // 1/sqrt(HEAD_DIM)
    v8f acc[8] = {};
    float mi[8], li[8];
#pragma unroll
    for (int r = 0; r < 8; ++r) { mi[r] = -1e30f; li[r] = 0.0f; }

    const int col = lane & 15;
    const int hi  = lane >> 4;

    stage(0, 0);
    wait_async0();
    __syncthreads();

    int buf = 0;
    for (int kv0 = 0; kv0 < SEQ; kv0 += 32) {
        if (kv0 + 32 < SEQ) stage(buf ^ 1, kv0 + 32);

        // ---- scores: S[16x32] = Q * K^T; batch all 8 K-fragments, then 8 WMMAs ----
        FragAB kf[8];
#pragma unroll
        for (int kk = 0; kk < 4; ++kk) {
            kf[2 * kk]     = lds_fragB(sK[buf], HEAD_DIM, 0,  kk * 32, lane);
            kf[2 * kk + 1] = lds_fragB(sK[buf], HEAD_DIM, 16, kk * 32, lane);
        }
        v8f s0 = {}, s1 = {};
#pragma unroll
        for (int kk = 0; kk < 4; ++kk) {
            s0 = wmma_bf16(qf[kk], kf[2 * kk],     s0);
            s1 = wmma_bf16(qf[kk], kf[2 * kk + 1], s1);
        }

        // ---- online softmax (row stats via half-wave shfl_xor reductions) ----
        float p0[8], p1[8], alpha[8];
#pragma unroll
        for (int r = 0; r < 8; ++r) {
            const int m = m0 + hi * 8 + r;
            float v0 = s0[r] * scale + mask[(size_t)m * SEQ + kv0 + col];
            float v1 = s1[r] * scale + mask[(size_t)m * SEQ + kv0 + 16 + col];
            float mx = fmaxf(v0, v1);
            mx = fmaxf(mx, __shfl_xor(mx, 1));
            mx = fmaxf(mx, __shfl_xor(mx, 2));
            mx = fmaxf(mx, __shfl_xor(mx, 4));
            mx = fmaxf(mx, __shfl_xor(mx, 8));
            const float mnew = fmaxf(mi[r], mx);
            const float a  = __expf(mi[r] - mnew);
            const float e0 = __expf(v0 - mnew);
            const float e1 = __expf(v1 - mnew);
            float rs = e0 + e1;
            rs += __shfl_xor(rs, 1);
            rs += __shfl_xor(rs, 2);
            rs += __shfl_xor(rs, 4);
            rs += __shfl_xor(rs, 8);
            li[r] = li[r] * a + rs;
            mi[r] = mnew;
            alpha[r] = a;
            p0[r] = e0; p1[r] = e1;
        }
#pragma unroll
        for (int tt = 0; tt < 8; ++tt)
#pragma unroll
            for (int r = 0; r < 8; ++r) acc[tt][r] = acc[tt][r] * alpha[r];

        // ---- re-layout P (C-layout -> A-fragment) through per-wave LDS tile ----
#pragma unroll
        for (int r = 0; r < 8; ++r) {
            const int m = hi * 8 + r;
            pshare[m * 32 + col]      = (bf16)p0[r];
            pshare[m * 32 + 16 + col] = (bf16)p1[r];
        }
        wait_ds0();
        FragAB pf;
        {
            const bf16* pp = pshare + (lane & 15) * 32 + hi * 8;
            pf.q[0] = *(const uint4*)(pp);
            pf.q[1] = *(const uint4*)(pp + 16);
        }

        // ---- O += P * V; batch all 8 V-fragments, then 8 WMMAs ----
        FragAB vf[8];
#pragma unroll
        for (int tt = 0; tt < 8; ++tt)
            vf[tt] = lds_fragB(sV[buf], 32, tt * 16, 0, lane);
#pragma unroll
        for (int tt = 0; tt < 8; ++tt)
            acc[tt] = wmma_bf16(pf, vf[tt], acc[tt]);

        wait_async0();
        __syncthreads();
        buf ^= 1;
    }

    // ---- epilogue: normalize, store O as bf16 [B,S,DIM] ----
#pragma unroll
    for (int r = 0; r < 8; ++r) li[r] = 1.0f / li[r];
#pragma unroll
    for (int tt = 0; tt < 8; ++tt)
#pragma unroll
        for (int r = 0; r < 8; ++r) {
            const int m = m0 + hi * 8 + r;
            const int d = tt * 16 + col;
            O[((size_t)(b * SEQ) + m) * DIMN + h * HEAD_DIM + d] = (bf16)(acc[tt][r] * li[r]);
        }
}

extern "C" void kernel_launch(void* const* d_in, const int* in_sizes, int n_in,
                              void* d_out, int out_size, void* d_ws, size_t ws_size,
                              hipStream_t stream) {
    const float* x    = (const float*)d_in[0];
    const float* mask = (const float*)d_in[1];
    const float* wq   = (const float*)d_in[2];
    const float* wk   = (const float*)d_in[3];
    const float* wv   = (const float*)d_in[4];
    const float* wo   = (const float*)d_in[5];
    float* out = (float*)d_out;

    // workspace carve-up (bf16 elements)
    bf16* ws = (bf16*)d_ws;
    const size_t NX = (size_t)BATCH * SEQ * DIMN;   // 8M elems
    const size_t NW = (size_t)DIMN * DIMN;          // 4M elems
    bf16* xb  = ws;
    bf16* wqb = xb  + NX;
    bf16* wkb = wqb + NW;
    bf16* wvb = wkb + NW;
    bf16* wob = wvb + NW;
    bf16* Qb  = wob + NW;
    bf16* Kb  = Qb  + NX;
    bf16* Vtb = Kb  + NX;
    bf16* Ob  = Vtb + NX;

    // 1) fp32 -> bf16 casts
    cast_f32_bf16<<<(int)(NX / 4 / 256), 256, 0, stream>>>(x, xb, (int)NX);
    cast_f32_bf16<<<(int)(NW / 4 / 256), 256, 0, stream>>>(wq, wqb, (int)NW);
    cast_f32_bf16<<<(int)(NW / 4 / 256), 256, 0, stream>>>(wk, wkb, (int)NW);
    cast_f32_bf16<<<(int)(NW / 4 / 256), 256, 0, stream>>>(wv, wvb, (int)NW);
    cast_f32_bf16<<<(int)(NW / 4 / 256), 256, 0, stream>>>(wo, wob, (int)NW);

    // 2) QKV projections: block tile 64x256 -> (4096/64)*(2048/256) = 512 blocks
    const int gblocks = (BATCH * SEQ / 64) * (DIMN / 256);
    gemm_xwT<<<gblocks, 256, 0, stream>>>(xb, wqb, nullptr, Qb,  1);
    gemm_xwT<<<gblocks, 256, 0, stream>>>(xb, wkb, nullptr, Kb,  1);
    gemm_xwT<<<gblocks, 256, 0, stream>>>(xb, wvb, nullptr, Vtb, 2);

    // 3) fused attention: 4096 wave tasks, 4 waves / 128-thread block
    const int tasks = BATCH * HEADS * (SEQ / 16);           // 4096
    flash_attn<<<tasks / 4, 128, 0, stream>>>(Qb, Kb, Vtb, mask, Ob);

    // 4) output projection -> fp32 d_out
    gemm_xwT<<<gblocks, 256, 0, stream>>>(Ob, wob, out, nullptr, 0);
}
